// DGCNN_82660940579283
// MI455X (gfx1250) — matrix operand order repaired and checked by reading, not compiled, not run on hardware
//
#include <hip/hip_runtime.h>
#include <hip/hip_bf16.h>
#include <cstddef>
#include <cstdint>

// DGCNN forward for MI455X (gfx1250), wave32 + WMMA f16->f32 + async global->LDS.
// Algebra: edge conv h[o,n,k] = V[o, idx[n,k]] + U[o,n] with V=Wl*X, U=(Wr-Wl)*X.
// BN scale > 0 and lrelu monotone => max over k (and over n for W4 stage) commutes
// past lrelu(bn(.)), so we max first, then apply the affine + lrelu once.
//
// Workspace layout (~27 MB): xc3 | idx | xcat(f16, 256ch) | V | U | xx | h4

#define B_ 8
#define N_ 2048
#define K_ 20
#define EPS_ 1e-5f
#define SLOPE_ 0.2f

typedef __attribute__((ext_vector_type(16))) _Float16 v16h;
typedef __attribute__((ext_vector_type(8)))  float    v8f;

__device__ __forceinline__ float lrelu(float x) { return x >= 0.f ? x : SLOPE_ * x; }

// A-matrix 16-bit 16x32 layout (ISA 7.12.2): element e of v16h, half = lane>>4
__device__ __forceinline__ int a_kmap(int e, int half) {
  return e + (e < 8 ? (half ? 8 : 0) : (half ? 16 : 8));
}

__device__ __forceinline__ v8f wmma_f16(v16h a, v16h b, v8f c) {
  return __builtin_amdgcn_wmma_f32_16x16x32_f16(false, a, false, b, (short)0, c,
                                                false, false);
}

// ---------------------------------------------------------------- prep: (B,N,3)->(B,3,N)
__global__ void prep_kernel(const float* __restrict__ x, float* __restrict__ xc3) {
  int gid = blockIdx.x * blockDim.x + threadIdx.x;
  if (gid >= B_ * N_) return;
  int n = gid % N_, b = gid / N_;
#pragma unroll
  for (int c = 0; c < 3; ++c)
    xc3[((size_t)b * 3 + c) * N_ + n] = x[((size_t)b * N_ + n) * 3 + c];
}

// ---------------------------------------------------------------- conv1 U/V (C=3, O=64)
__global__ void conv1_uv_kernel(const float* __restrict__ W1, const float* __restrict__ xc3,
                                float* __restrict__ V, float* __restrict__ U) {
  size_t gid = (size_t)blockIdx.x * blockDim.x + threadIdx.x;
  if (gid >= (size_t)B_ * 64 * N_) return;
  int n = gid % N_;
  int o = (gid / N_) % 64;
  int b = (int)(gid / ((size_t)N_ * 64));
  const float* Xb = xc3 + (size_t)b * 3 * N_;
  float v = 0.f, u = 0.f;
#pragma unroll
  for (int c = 0; c < 3; ++c) {
    float xv = Xb[(size_t)c * N_ + n];
    float wl = W1[o * 6 + c], wr = W1[o * 6 + 3 + c];
    v += wl * xv;
    u += (wr - wl) * xv;
  }
  V[gid] = v;
  U[gid] = u;
}

// ---------------------------------------------------------------- KNN, C=3 (block 1)
// 8 waves/WG, one row per wave; scores in dynamic LDS (8*N_ floats = 64 KB).
__global__ void knn3_kernel(const float* __restrict__ xc3, int* __restrict__ idx) {
  extern __shared__ float smem[];
  const int wave = threadIdx.x >> 5;
  const int lane = threadIdx.x & 31;
  const int b = blockIdx.x / (N_ / 8);
  const int n = (blockIdx.x % (N_ / 8)) * 8 + wave;
  const float* Xb = xc3 + (size_t)b * 3 * N_;
  float x0 = Xb[n], x1 = Xb[N_ + n], x2 = Xb[2 * N_ + n];
  float* srow = smem + wave * N_;
  for (int m = lane; m < N_; m += 32) {
    float y0 = Xb[m], y1 = Xb[N_ + m], y2 = Xb[2 * N_ + m];
    float d = x0 * y0 + x1 * y1 + x2 * y2;
    srow[m] = 2.f * d - (y0 * y0 + y1 * y1 + y2 * y2);  // -xx[n] const: argmax-invariant
  }
  __syncthreads();
  for (int kk = 0; kk < K_; ++kk) {
    float best = -3.4e38f;
    int bi = N_;
    for (int m = lane; m < N_; m += 32) {
      float v = srow[m];
      if (v > best || (v == best && m < bi)) { best = v; bi = m; }
    }
#pragma unroll
    for (int off = 16; off > 0; off >>= 1) {
      float ov = __shfl_xor(best, off);
      int oi = __shfl_xor(bi, off);
      if (ov > best || (ov == best && oi < bi)) { best = ov; bi = oi; }
    }
    if (lane == 0) {
      idx[((size_t)b * N_ + n) * K_ + kk] = bi;
      srow[bi] = -3.4e38f;
    }
    __syncthreads();
  }
}

// ---------------------------------------------------------------- xx = colwise ||.||^2 of f16 feature block
__global__ void xx_kernel(const _Float16* __restrict__ xcat, int cbase, int C,
                          float* __restrict__ xx) {
  int gid = blockIdx.x * blockDim.x + threadIdx.x;
  if (gid >= B_ * N_) return;
  int n = gid % N_, b = gid / N_;
  const _Float16* X = xcat + ((size_t)b * 256 + cbase) * N_;
  float s = 0.f;
  for (int c = 0; c < C; ++c) {
    float v = (float)X[(size_t)c * N_ + n];
    s += v * v;
  }
  xx[gid] = s;
}

// ---------------------------------------------------------------- KNN via WMMA, C=64 (blocks 2,3)
// One WG (8 waves) per 16-row tile: scores (16 x 2048 f32 = 128 KB dyn LDS) via
// v_wmma_f32_16x16x32_f16 inner products; then per-wave iterative top-K.
__global__ void knn_wmma_kernel(const _Float16* __restrict__ xcat, int cbase,
                                const float* __restrict__ xx, int* __restrict__ idx) {
  extern __shared__ float smem[];  // 16 * N_
  const int tiles_n = N_ / 16;
  const int b = blockIdx.x / tiles_n;
  const int n0 = (blockIdx.x % tiles_n) * 16;
  const int wave = threadIdx.x >> 5;
  const int lane = threadIdx.x & 31;
  const int half = lane >> 4;
  const int r = lane & 15;
  const _Float16* X = xcat + ((size_t)b * 256 + cbase) * N_;

  // A fragments for rows n0..n0+15 over K=C=64 (2 fragments of K=32)
  v16h afr[2];
#pragma unroll
  for (int kk = 0; kk < 2; ++kk) {
    v16h a;
#pragma unroll
    for (int e = 0; e < 16; ++e) {
      int k = kk * 32 + a_kmap(e, half);
      a[e] = X[(size_t)k * N_ + n0 + r];
    }
    afr[kk] = a;
  }
  for (int t = wave; t < N_ / 16; t += 8) {
    int m0 = t * 16;
    v8f acc = {};
#pragma unroll
    for (int kk = 0; kk < 2; ++kk) {
      v16h bb;
#pragma unroll
      for (int e = 0; e < 16; ++e) {
        int k = kk * 32 + a_kmap(e, half);
        bb[e] = X[(size_t)k * N_ + m0 + r];
      }
      acc = wmma_f16(afr[kk], bb, acc);
    }
#pragma unroll
    for (int e = 0; e < 8; ++e) {
      int M = e + half * 8;
      int col = m0 + r;
      smem[M * N_ + col] = 2.f * acc[e] - xx[(size_t)b * N_ + col];
    }
  }
  __syncthreads();
  // top-K: each wave owns rows 2*wave, 2*wave+1 (uniform loop counts -> barriers legal)
  for (int rr = 0; rr < 2; ++rr) {
    int row = wave * 2 + rr;
    float* srow = smem + row * N_;
    int n = n0 + row;
    for (int kk = 0; kk < K_; ++kk) {
      float best = -3.4e38f;
      int bi = N_;
      for (int m = lane; m < N_; m += 32) {
        float v = srow[m];
        if (v > best || (v == best && m < bi)) { best = v; bi = m; }
      }
#pragma unroll
      for (int off = 16; off > 0; off >>= 1) {
        float ov = __shfl_xor(best, off);
        int oi = __shfl_xor(bi, off);
        if (ov > best || (ov == best && oi < bi)) { best = ov; bi = oi; }
      }
      if (lane == 0) {
        idx[((size_t)b * N_ + n) * K_ + kk] = bi;
        srow[bi] = -3.4e38f;
      }
      __syncthreads();
    }
  }
}

// ---------------------------------------------------------------- fused WMMA GEMM pair:
//   V = W[:,0:64] @ X,  U = (W[:,64:128] - W[:,0:64]) @ X   (C=64, ldw=128)
// One wave per 16x16 output tile. The 64x16 f16 X tile (2 KB) is staged into LDS
// with GLOBAL_LOAD_ASYNC_TO_LDS_B128 (ASYNCcnt) and consumed by both WMMA chains.
__global__ void gemm_uv_kernel(const float* __restrict__ W,
                               const _Float16* __restrict__ xcat, int cbase, int O,
                               float* __restrict__ V, float* __restrict__ U) {
  __shared__ _Float16 stage[64 * 16];  // k-major: stage[k*16 + n]; LDS offset 0
  const int n0 = blockIdx.x * 16;
  const int o0 = blockIdx.y * 16;
  const int b = blockIdx.z;
  const int l = threadIdx.x;  // 32 threads, EXEC all-ones
  const int half = l >> 4;
  const int r = l & 15;
  const _Float16* X = xcat + ((size_t)b * 256 + cbase) * N_;

  // lane l, issue i: global row k = i*16 + (l>>1), 8 halves at col n0 + (l&1)*8
  //                  -> LDS byte offset i*512 + l*16  (== (k*16 + (l&1)*8)*2)
#pragma unroll
  for (int i = 0; i < 4; ++i) {
    int k = i * 16 + (l >> 1);
    uint64_t ga = (uint64_t)(uintptr_t)(X + (size_t)k * N_ + n0 + (l & 1) * 8);
    uint32_t loff = (uint32_t)(i * 512 + l * 16);
    asm volatile("global_load_async_to_lds_b128 %0, %1, off"
                 :: "v"(loff), "v"(ga) : "memory");
  }
  asm volatile("s_wait_asynccnt 0x0" ::: "memory");

  v8f accv = {}, accu = {};
#pragma unroll
  for (int kb = 0; kb < 64; kb += 32) {
    v16h av, au, bb;
#pragma unroll
    for (int e = 0; e < 16; ++e) {
      int k = kb + a_kmap(e, half);
      float wl = W[(size_t)(o0 + r) * 128 + k];
      float wr = W[(size_t)(o0 + r) * 128 + 64 + k];
      av[e] = (_Float16)wl;
      au[e] = (_Float16)(wr - wl);
      int k2 = kb + a_kmap(e, half);
      bb[e] = stage[k2 * 16 + r];
    }
    accv = wmma_f16(av, bb, accv);
    accu = wmma_f16(au, bb, accu);
  }
#pragma unroll
  for (int e = 0; e < 8; ++e) {
    int M = e + half * 8;
    V[((size_t)b * O + o0 + M) * N_ + n0 + r] = accv[e];
    U[((size_t)b * O + o0 + M) * N_ + n0 + r] = accu[e];
  }
}

// ---------------------------------------------------------------- gather + max_k + BN + lrelu -> f16 feature
__global__ void gather_max_kernel(const float* __restrict__ V, const float* __restrict__ U,
                                  const int* __restrict__ idx, int O,
                                  const float* __restrict__ g, const float* __restrict__ bb,
                                  const float* __restrict__ mm, const float* __restrict__ vv,
                                  _Float16* __restrict__ xcat, int cbase) {
  size_t gid = (size_t)blockIdx.x * blockDim.x + threadIdx.x;
  if (gid >= (size_t)B_ * O * N_) return;
  int n = gid % N_;
  int o = (gid / N_) % O;
  int b = (int)(gid / ((size_t)N_ * O));
  const float* Vrow = V + ((size_t)b * O + o) * N_;
  const int* irow = idx + ((size_t)b * N_ + n) * K_;
  float m = -3.4e38f;
#pragma unroll 4
  for (int k = 0; k < K_; ++k) m = fmaxf(m, Vrow[irow[k]]);
  float h = m + U[((size_t)b * O + o) * N_ + n];
  float y = g[o] * (h - mm[o]) * rsqrtf(vv[o] + EPS_) + bb[o];
  xcat[((size_t)b * 256 + cbase + o) * N_ + n] = (_Float16)lrelu(y);
}

// ---------------------------------------------------------------- W4 GEMM (1024x256 @ 256xN) + max over N + BN + lrelu
__global__ void gemm_w4_max_kernel(const float* __restrict__ W4,
                                   const _Float16* __restrict__ xcat,
                                   const float* __restrict__ g, const float* __restrict__ bb,
                                   const float* __restrict__ mm, const float* __restrict__ vv,
                                   float* __restrict__ h4) {
  const int o0 = blockIdx.x * 16;
  const int b = blockIdx.y;
  const int lane = threadIdx.x & 31;
  const int half = lane >> 4;
  const int r = lane & 15;
  const _Float16* X = xcat + (size_t)b * 256 * N_;
  v16h afr[8];
#pragma unroll
  for (int kk = 0; kk < 8; ++kk) {
    v16h a;
#pragma unroll
    for (int e = 0; e < 16; ++e) {
      int k = kk * 32 + a_kmap(e, half);
      a[e] = (_Float16)W4[(size_t)(o0 + r) * 256 + k];
    }
    afr[kk] = a;
  }
  v8f vmax;
#pragma unroll
  for (int e = 0; e < 8; ++e) vmax[e] = -3.4e38f;
  for (int t = 0; t < N_ / 16; ++t) {
    int n0 = t * 16;
    v8f acc = {};
#pragma unroll
    for (int kk = 0; kk < 8; ++kk) {
      v16h bbf;
#pragma unroll
      for (int e = 0; e < 16; ++e) {
        int k = kk * 32 + a_kmap(e, half);
        bbf[e] = X[(size_t)k * N_ + n0 + r];
      }
      acc = wmma_f16(afr[kk], bbf, acc);
    }
#pragma unroll
    for (int e = 0; e < 8; ++e) vmax[e] = fmaxf(vmax[e], acc[e]);
  }
#pragma unroll
  for (int e = 0; e < 8; ++e) {
    float v = vmax[e];
#pragma unroll
    for (int off = 1; off < 16; off <<= 1) v = fmaxf(v, __shfl_xor(v, off));
    if (r == 0) {
      int o = o0 + e + half * 8;
      float y = g[o] * (v - mm[o]) * rsqrtf(vv[o] + EPS_) + bb[o];
      h4[(size_t)b * 1024 + o] = lrelu(y);
    }
  }
}

// ---------------------------------------------------------------- MLP head (tiny): 1024->512->256->2
__global__ void head_kernel(const float* __restrict__ h4,
                            const float* __restrict__ l1w, const float* __restrict__ l1b,
                            const float* __restrict__ g5, const float* __restrict__ b5,
                            const float* __restrict__ m5, const float* __restrict__ v5,
                            const float* __restrict__ l2w, const float* __restrict__ l2b,
                            const float* __restrict__ g6, const float* __restrict__ b6,
                            const float* __restrict__ m6, const float* __restrict__ v6,
                            const float* __restrict__ l3w, const float* __restrict__ l3b,
                            float* __restrict__ out) {
  __shared__ float sin_[1024], s1[512], s2[256];
  int tid = threadIdx.x;  // 256 threads
  for (int b = 0; b < B_; ++b) {
    for (int i = tid; i < 1024; i += 256) sin_[i] = h4[b * 1024 + i];
    __syncthreads();
    for (int o = tid; o < 512; o += 256) {
      float acc = l1b[o];
      for (int c = 0; c < 1024; ++c) acc += l1w[o * 1024 + c] * sin_[c];
      float y = g5[o] * (acc - m5[o]) * rsqrtf(v5[o] + EPS_) + b5[o];
      s1[o] = lrelu(y);
    }
    __syncthreads();
    if (tid < 256) {
      int o = tid;
      float acc = l2b[o];
      for (int c = 0; c < 512; ++c) acc += l2w[o * 512 + c] * s1[c];
      float y = g6[o] * (acc - m6[o]) * rsqrtf(v6[o] + EPS_) + b6[o];
      s2[o] = lrelu(y);
    }
    __syncthreads();
    if (tid < 2) {
      float acc = l3b[tid];
      for (int c = 0; c < 256; ++c) acc += l3w[tid * 256 + c] * s2[c];
      out[b * 2 + tid] = acc;
    }
    __syncthreads();
  }
}

// ----------------------------------------------------------------
extern "C" void kernel_launch(void* const* d_in, const int* in_sizes, int n_in,
                              void* d_out, int out_size, void* d_ws, size_t ws_size,
                              hipStream_t stream) {
  (void)in_sizes; (void)n_in; (void)out_size; (void)ws_size;
  const float* x   = (const float*)d_in[0];
  const float* W1  = (const float*)d_in[1];
  const float* bn1g = (const float*)d_in[2],  *bn1b = (const float*)d_in[3];
  const float* bn1m = (const float*)d_in[4],  *bn1v = (const float*)d_in[5];
  const float* W2  = (const float*)d_in[6];
  const float* bn2g = (const float*)d_in[7],  *bn2b = (const float*)d_in[8];
  const float* bn2m = (const float*)d_in[9],  *bn2v = (const float*)d_in[10];
  const float* W3  = (const float*)d_in[11];
  const float* bn3g = (const float*)d_in[12], *bn3b = (const float*)d_in[13];
  const float* bn3m = (const float*)d_in[14], *bn3v = (const float*)d_in[15];
  const float* W4  = (const float*)d_in[16];
  const float* bn4g = (const float*)d_in[17], *bn4b = (const float*)d_in[18];
  const float* bn4m = (const float*)d_in[19], *bn4v = (const float*)d_in[20];
  const float* l1w = (const float*)d_in[21], *l1b = (const float*)d_in[22];
  const float* bn5g = (const float*)d_in[23], *bn5b = (const float*)d_in[24];
  const float* bn5m = (const float*)d_in[25], *bn5v = (const float*)d_in[26];
  const float* l2w = (const float*)d_in[27], *l2b = (const float*)d_in[28];
  const float* bn6g = (const float*)d_in[29], *bn6b = (const float*)d_in[30];
  const float* bn6m = (const float*)d_in[31], *bn6v = (const float*)d_in[32];
  const float* l3w = (const float*)d_in[33], *l3b = (const float*)d_in[34];
  float* out = (float*)d_out;

  // workspace partition (aligned to 256 B)
  uintptr_t p = (uintptr_t)d_ws;
  auto take = [&](size_t bytes) {
    uintptr_t q = (p + 255) & ~(uintptr_t)255;
    p = q + bytes;
    return (void*)q;
  };
  float*    xc3  = (float*)take(sizeof(float) * B_ * 3 * N_);
  int*      idx  = (int*)take(sizeof(int) * (size_t)B_ * N_ * K_);
  _Float16* xcat = (_Float16*)take(sizeof(_Float16) * (size_t)B_ * 256 * N_);
  float*    V    = (float*)take(sizeof(float) * (size_t)B_ * 128 * N_);
  float*    U    = (float*)take(sizeof(float) * (size_t)B_ * 128 * N_);
  float*    xx   = (float*)take(sizeof(float) * (size_t)B_ * N_);
  float*    h4   = (float*)take(sizeof(float) * (size_t)B_ * 1024);

  const int T = 256;
  // block 1 (C=3)
  prep_kernel<<<(B_ * N_ + T - 1) / T, T, 0, stream>>>(x, xc3);
  conv1_uv_kernel<<<((size_t)B_ * 64 * N_ + T - 1) / T, T, 0, stream>>>(W1, xc3, V, U);
  knn3_kernel<<<B_ * N_ / 8, T, 8 * N_ * sizeof(float), stream>>>(xc3, idx);
  gather_max_kernel<<<((size_t)B_ * 64 * N_ + T - 1) / T, T, 0, stream>>>(
      V, U, idx, 64, bn1g, bn1b, bn1m, bn1v, xcat, 0);
  // block 2 (C=64 -> O=64)
  xx_kernel<<<(B_ * N_ + T - 1) / T, T, 0, stream>>>(xcat, 0, 64, xx);
  knn_wmma_kernel<<<B_ * (N_ / 16), T, 16 * N_ * sizeof(float), stream>>>(xcat, 0, xx, idx);
  gemm_uv_kernel<<<dim3(N_ / 16, 64 / 16, B_), 32, 0, stream>>>(W2, xcat, 0, 64, V, U);
  gather_max_kernel<<<((size_t)B_ * 64 * N_ + T - 1) / T, T, 0, stream>>>(
      V, U, idx, 64, bn2g, bn2b, bn2m, bn2v, xcat, 64);
  // block 3 (C=64 -> O=128)
  xx_kernel<<<(B_ * N_ + T - 1) / T, T, 0, stream>>>(xcat, 64, 64, xx);
  knn_wmma_kernel<<<B_ * (N_ / 16), T, 16 * N_ * sizeof(float), stream>>>(xcat, 64, xx, idx);
  gemm_uv_kernel<<<dim3(N_ / 16, 128 / 16, B_), 32, 0, stream>>>(W3, xcat, 64, 128, V, U);
  gather_max_kernel<<<((size_t)B_ * 128 * N_ + T - 1) / T, T, 0, stream>>>(
      V, U, idx, 128, bn3g, bn3b, bn3m, bn3v, xcat, 128);
  // W4 + global max pool, then head
  gemm_w4_max_kernel<<<dim3(1024 / 16, B_), 32, 0, stream>>>(W4, xcat, bn4g, bn4b, bn4m, bn4v, h4);
  head_kernel<<<1, 256, 0, stream>>>(h4, l1w, l1b, bn5g, bn5b, bn5m, bn5v,
                                     l2w, l2b, bn6g, bn6b, bn6m, bn6v, l3w, l3b, out);
}
